// SVAT_89730456748203
// MI455X (gfx1250) — matrix operand order repaired and checked by reading, not compiled, not run on hardware
//
#include <hip/hip_runtime.h>
#include <hip/hip_bf16.h>

typedef __attribute__((ext_vector_type(16))) _Float16 v16h;
typedef __attribute__((ext_vector_type(8)))  _Float16 v8h;
typedef __attribute__((ext_vector_type(4)))  _Float16 h4;
typedef __attribute__((ext_vector_type(8)))  float    v8f;

#define NN   8192
#define TT   32
#define FF   16
#define HH   128
#define G3   384
#define NHEADS 4
#define EE   2048
#define NNZV 65536

// ---------------- WMMA helpers (wave32, 16x16x32 f16 -> f32) ----------------
__device__ __forceinline__ v8f wmma16(const v16h& a, const v16h& b, const v8f& c) {
  return __builtin_amdgcn_wmma_f32_16x16x32_f16(false, a, false, b, (short)0, c, false, false);
}
// A fragment (16xK slice, K0..K0+31): per lane two contiguous 16B runs.
// lanes 0-15: K [k0+0,8) and [k0+16,24); lanes 16-31: +8.
__device__ __forceinline__ v16h load_a_frag(const _Float16* base, int stride, int lane, int k0) {
  int r  = lane & 15;
  int kb = k0 + ((lane >> 4) << 3);
  const _Float16* p = base + r * stride + kb;
  v8h lo = *(const v8h*)p;
  v8h hi = *(const v8h*)(p + 16);
  return __builtin_shufflevector(lo, hi, 0,1,2,3,4,5,6,7,8,9,10,11,12,13,14,15);
}
// B fragment (Kx16 from row-major W[out][K]): per lane two contiguous 16B runs (same K order as A).
__device__ __forceinline__ v16h load_b_frag(const _Float16* base, int stride, int lane, int k0) {
  int c  = lane & 15;
  int kb = k0 + ((lane >> 4) << 3);
  const _Float16* p = base + c * stride + kb;
  v8h lo = *(const v8h*)p;
  v8h hi = *(const v8h*)(p + 16);
  return __builtin_shufflevector(lo, hi, 0,1,2,3,4,5,6,7,8,9,10,11,12,13,14,15);
}
__device__ __forceinline__ int c_row(int lane, int v){ return v + ((lane >> 4) << 3); }
__device__ __forceinline__ int c_col(int lane)       { return lane & 15; }

__device__ __forceinline__ void cvt_store4(_Float16* dst, float4 v) {
  h4 h; h[0] = (_Float16)v.x; h[1] = (_Float16)v.y; h[2] = (_Float16)v.z; h[3] = (_Float16)v.w;
  *(h4*)dst = h;
}

__device__ __forceinline__ unsigned fenc(float f) {
  unsigned u = __float_as_uint(f);
  return (u >> 31) ? ~u : (u | 0x80000000u);
}
__device__ __forceinline__ float fdec(unsigned u) {
  return __uint_as_float((u & 0x80000000u) ? (u ^ 0x80000000u) : ~u);
}
__device__ __forceinline__ float leaky(float x, float s) { return x > 0.f ? x : s * x; }
__device__ __forceinline__ float sigmoidf(float x) { return 1.f / (1.f + __expf(-x)); }

// ---------------- BatchNorm statistics ----------------
__global__ void bn_stats_kernel(const float* __restrict__ x, int rows, int cols,
                                float* __restrict__ mean, float* __restrict__ rstd) {
  int f = blockIdx.x, tid = threadIdx.x;
  __shared__ float s1[256], s2[256];
  float sum = 0.f, sq = 0.f;
  for (int n = tid; n < rows; n += 256) {
    float v = x[(size_t)n * cols + f];
    sum += v; sq += v * v;
  }
  s1[tid] = sum; s2[tid] = sq;
  __syncthreads();
  for (int s = 128; s > 0; s >>= 1) {
    if (tid < s) { s1[tid] += s1[tid + s]; s2[tid] += s2[tid + s]; }
    __syncthreads();
  }
  if (tid == 0) {
    float m = s1[0] / rows;
    float v = s2[0] / rows - m * m;
    mean[f] = m;
    rstd[f] = rsqrtf(v + 1e-5f);
  }
}

__global__ void bn_apply_kernel(const float* __restrict__ x, const float* __restrict__ mean,
                                const float* __restrict__ rstd, const float* __restrict__ g,
                                const float* __restrict__ b, float* __restrict__ y,
                                int total, int cols) {
  int idx = blockIdx.x * 256 + threadIdx.x;
  if (idx >= total) return;
  int f = idx % cols;
  y[idx] = (x[idx] - mean[f]) * rstd[f] * g[f] + b[f];
}

// ---------------- Fused GRU scan: 16 nodes / wave; all operands f16 in LDS ----------------
__global__ __launch_bounds__(32) void gru_kernel(
    const float* __restrict__ price, const float* __restrict__ mean1,
    const float* __restrict__ rstd1, const float* __restrict__ g1, const float* __restrict__ bb1,
    const float* __restrict__ W_ih, const float* __restrict__ W_hh,
    const float* __restrict__ b_ih, const float* __restrict__ b_hh,
    float* __restrict__ context, float* __restrict__ hT) {
  extern __shared__ char smem[];
  _Float16* sWhh = (_Float16*)smem;                         // 384x128 f16  (96KB)
  _Float16* sWih = (_Float16*)(smem + 98304);               // 384x32  f16, K padded (24KB)
  _Float16* sxt  = (_Float16*)(smem + 123904 - 1024);       // 16x32 f16 x_t tile (1KB) @122880
  _Float16* shh  = (_Float16*)(smem + 123904);              // 16x128 f16 h (4KB)
  float* sh  = (float*)(smem + 128000);                     // 16x128 f32 h (8KB)
  float* sxg = (float*)(smem + 136192);                     // 16x384 f32 (24KB)
  float* shg = (float*)(smem + 160768);                     // 16x384 f32 (24KB)

  int lane = threadIdx.x;
  int nb = blockIdx.x * 16;

  // stage weights: W_hh via vectorized float4 -> f16
  for (int i = lane; i < (G3 * HH) / 4; i += 32)
    cvt_store4(&sWhh[i * 4], ((const float4*)W_hh)[i]);
  // W_ih padded from K=16 to K=32
  for (int i = lane; i < G3 * 32; i += 32) {
    int row = i >> 5, k = i & 31;
    sWih[i] = (k < FF) ? (_Float16)W_ih[row * FF + k] : (_Float16)0.f;
  }
  for (int i = lane; i < 16 * HH; i += 32) { sh[i] = 0.f; shh[i] = (_Float16)0.f; }
  __syncthreads();

  for (int t = 0; t < TT; ++t) {
    // stage BN(x_t) tile into LDS f16 (padded to K=32)
    for (int idx = lane; idx < 16 * 32; idx += 32) {
      int r = idx >> 5, k = idx & 31;
      float xv = 0.f;
      if (k < FF) {
        int col = t * FF + k;
        float raw = price[(size_t)(nb + r) * (TT * FF) + col];
        xv = (raw - mean1[col]) * rstd1[col] * g1[col] + bb1[col];
      }
      sxt[idx] = (_Float16)xv;
    }
    // A fragments (reused across all 24 output tiles)
    v16h ax = load_a_frag(sxt, 32, lane, 0);
    v16h ah0 = load_a_frag(shh, HH, lane, 0);
    v16h ah1 = load_a_frag(shh, HH, lane, 32);
    v16h ah2 = load_a_frag(shh, HH, lane, 64);
    v16h ah3 = load_a_frag(shh, HH, lane, 96);

    for (int gt = 0; gt < G3 / 16; ++gt) {
      // xg tile
      v8f accx = {};
      accx = wmma16(ax, load_b_frag(sWih + gt * 16 * 32, 32, lane, 0), accx);
      // hg tile: 4 K-chunks
      v8f acch = {};
      const _Float16* wb = sWhh + gt * 16 * HH;
      acch = wmma16(ah0, load_b_frag(wb, HH, lane, 0),  acch);
      acch = wmma16(ah1, load_b_frag(wb, HH, lane, 32), acch);
      acch = wmma16(ah2, load_b_frag(wb, HH, lane, 64), acch);
      acch = wmma16(ah3, load_b_frag(wb, HH, lane, 96), acch);
      int col = gt * 16 + c_col(lane);
      float bi = b_ih[col], bh = b_hh[col];
      #pragma unroll
      for (int v = 0; v < 8; ++v) {
        sxg[c_row(lane, v) * G3 + col] = accx[v] + bi;
        shg[c_row(lane, v) * G3 + col] = acch[v] + bh;
      }
    }
    __syncthreads();
    // elementwise gate update, vectorized x4
    for (int vidx = lane; vidx < (16 * HH) / 4; vidx += 32) {
      int idx = vidx * 4;
      int rl = idx >> 7, j = idx & 127;
      float4 hv;
      float* hvp = (float*)&hv;
      #pragma unroll
      for (int u = 0; u < 4; ++u) {
        int jj = j + u;
        float r = sigmoidf(sxg[rl * G3 + jj]          + shg[rl * G3 + jj]);
        float z = sigmoidf(sxg[rl * G3 + HH + jj]     + shg[rl * G3 + HH + jj]);
        float n = tanhf(sxg[rl * G3 + 2 * HH + jj] + r * shg[rl * G3 + 2 * HH + jj]);
        hvp[u] = (1.f - z) * n + z * sh[idx + u];
        sh[idx + u] = hvp[u];
      }
      cvt_store4(&shh[idx], hv);
      *(float4*)&context[((size_t)(nb + rl) * TT + t) * HH + j] = hv;  // stays in 192MB L2
    }
    __syncthreads();
  }
  for (int vidx = lane; vidx < (16 * HH) / 4; vidx += 32) {
    int idx = vidx * 4;
    *(float4*)&hT[(size_t)(nb + (idx >> 7)) * HH + (idx & 127)] = *(float4*)&sh[idx];
  }
}

// ---------------- Fused temporal attention ----------------
__global__ __launch_bounds__(32) void attn_kernel(
    const float* __restrict__ hT, const float* __restrict__ context,
    const float* __restrict__ Wq, const float* __restrict__ Wout,
    const float* __restrict__ ae, const float* __restrict__ ab, float* __restrict__ nf) {
  __shared__ _Float16 sht[16 * HH];      // hT tile f16
  __shared__ _Float16 sB[16 * 256];      // weight tile staging f16
  __shared__ _Float16 sqh[16 * HH];      // q f16
  __shared__ _Float16 smixh[16 * HH];    // mixsum f16
  __shared__ float sq[16 * HH];          // q f32 (for scores)
  __shared__ float sw[16 * TT];          // softmax weights
  int lane = threadIdx.x;
  int nb = blockIdx.x * 16;

  // stage hT tile (contiguous 2048 floats)
  for (int i = lane; i < (16 * HH) / 4; i += 32)
    cvt_store4(&sht[i * 4], ((const float4*)(hT + (size_t)nb * HH))[i]);

  v16h a0 = load_a_frag(sht, HH, lane, 0);
  v16h a1 = load_a_frag(sht, HH, lane, 32);
  v16h a2 = load_a_frag(sht, HH, lane, 64);
  v16h a3 = load_a_frag(sht, HH, lane, 96);

  // q = hT @ Wq^T, tile-by-tile with staged weights
  for (int ot = 0; ot < HH / 16; ++ot) {
    const float4* wsrc = (const float4*)(Wq + (size_t)ot * 16 * HH);  // 16 contiguous rows
    for (int i = lane; i < (16 * HH) / 4; i += 32) cvt_store4(&sB[i * 4], wsrc[i]);
    v8f acc = {};
    acc = wmma16(a0, load_b_frag(sB, HH, lane, 0),  acc);
    acc = wmma16(a1, load_b_frag(sB, HH, lane, 32), acc);
    acc = wmma16(a2, load_b_frag(sB, HH, lane, 64), acc);
    acc = wmma16(a3, load_b_frag(sB, HH, lane, 96), acc);
    int col = ot * 16 + c_col(lane);
    #pragma unroll
    for (int v = 0; v < 8; ++v) {
      int r = c_row(lane, v);
      sq[r * HH + col]  = acc[v];
      sqh[r * HH + col] = (_Float16)acc[v];
    }
  }
  __syncthreads();

  // scores + softmax over T (one node per lane 0..15)
  if (lane < 16) {
    float sc[TT];
    float m = -1e30f;
    for (int t = 0; t < TT; ++t) {
      float s = 0.f;
      const float* ctx = context + ((size_t)(nb + lane) * TT + t) * HH;
      for (int c = 0; c < HH; ++c) s += sq[lane * HH + c] * ctx[c];
      sc[t] = s;
      m = fmaxf(m, s);
    }
    float sum = 0.f;
    for (int t = 0; t < TT; ++t) { sc[t] = __expf(sc[t] - m); sum += sc[t]; }
    for (int t = 0; t < TT; ++t) sw[lane * TT + t] = sc[t] / sum;
  }
  __syncthreads();

  // mixsum[n,h] = sum_t relu(ae*mix*exp(-ab*dt)) + mix
  for (int idx = lane; idx < 16 * HH; idx += 32) {
    int nl = idx >> 7, hh = idx & 127;
    float aev = ae[nb + nl], abv = ab[nb + nl];
    float s = 0.f;
    for (int t = 0; t < TT; ++t) {
      float c = context[((size_t)(nb + nl) * TT + t) * HH + hh];
      float mix = sw[nl * TT + t] * c;
      float bt = __expf(-abv * (float)(TT - 1 - t));
      s += fmaxf(aev * mix * bt, 0.f) + mix;
    }
    smixh[idx] = (_Float16)s;
  }
  __syncthreads();

  // nf = tanh([mixsum, q] @ Wout^T)   (K=256)
  v16h af[8];
  #pragma unroll
  for (int kc = 0; kc < 8; ++kc)
    af[kc] = load_a_frag((kc < 4) ? smixh : sqh, HH, lane, (kc & 3) * 32);

  for (int ot = 0; ot < HH / 16; ++ot) {
    const float4* wsrc = (const float4*)(Wout + (size_t)ot * 16 * 256);
    for (int i = lane; i < (16 * 256) / 4; i += 32) cvt_store4(&sB[i * 4], wsrc[i]);
    v8f acc = {};
    #pragma unroll
    for (int kc = 0; kc < 8; ++kc)
      acc = wmma16(af[kc], load_b_frag(sB, 256, lane, kc * 32), acc);
    int col = ot * 16 + c_col(lane);
    #pragma unroll
    for (int v = 0; v < 8; ++v)
      nf[(size_t)(nb + c_row(lane, v)) * HH + col] = tanhf(acc[v]);
  }
}

// ---------------- WMMA GEMM: C[M,Nout] = A[M,K=128] @ B[Nout,K=128]^T ----------------
// block = 4 waves; 16-row M tile x 64-col N tile per block (16 cols per wave)
__global__ __launch_bounds__(128) void gemm_wmma_kernel(
    const float* __restrict__ A, const float* __restrict__ B, float* __restrict__ C,
    int Nout) {
  __shared__ _Float16 As[16 * HH];
  __shared__ _Float16 Bs[64 * HH];
  int tid = threadIdx.x;
  int lane = tid & 31, w = tid >> 5;
  int m0 = blockIdx.x * 16;
  int n0g = blockIdx.y * 64;

  const float4* asrc = (const float4*)(A + (size_t)m0 * HH);
  for (int i = tid; i < (16 * HH) / 4; i += 128) cvt_store4(&As[i * 4], asrc[i]);
  const float4* bsrc = (const float4*)(B + (size_t)n0g * HH);
  for (int i = tid; i < (64 * HH) / 4; i += 128) cvt_store4(&Bs[i * 4], bsrc[i]);
  __syncthreads();

  v16h a0 = load_a_frag(As, HH, lane, 0);
  v16h a1 = load_a_frag(As, HH, lane, 32);
  v16h a2 = load_a_frag(As, HH, lane, 64);
  v16h a3 = load_a_frag(As, HH, lane, 96);
  const _Float16* wb = Bs + w * 16 * HH;
  v8f acc = {};
  acc = wmma16(a0, load_b_frag(wb, HH, lane, 0),  acc);
  acc = wmma16(a1, load_b_frag(wb, HH, lane, 32), acc);
  acc = wmma16(a2, load_b_frag(wb, HH, lane, 64), acc);
  acc = wmma16(a3, load_b_frag(wb, HH, lane, 96), acc);

  int col = n0g + w * 16 + c_col(lane);
  #pragma unroll
  for (int v = 0; v < 8; ++v)
    C[(size_t)(m0 + c_row(lane, v)) * Nout + col] = acc[v];
}

// ---------------- Hypergraph scatter/segment kernels ----------------
__global__ void edge_feat_kernel(const float* __restrict__ x, const int* __restrict__ src,
                                 const int* __restrict__ dst, float* __restrict__ ef, int nnz) {
  int idx = blockIdx.x * 256 + threadIdx.x;
  if (idx >= nnz * HH) return;
  int i = idx >> 7, f = idx & 127;
  atomicAdd(&ef[(size_t)dst[i] * HH + f], x[(size_t)src[i] * HH + f]);
}

__global__ void att_score_kernel(const float* __restrict__ xw, const float* __restrict__ att,
                                 float* __restrict__ s, int rows, int heads, int attOff) {
  int idx = blockIdx.x * 256 + threadIdx.x;
  if (idx >= rows * heads) return;
  int r = idx / heads, h = idx % heads;
  const float* xp = xw + ((size_t)r * heads + h) * HH;
  const float* ap = att + h * (2 * HH) + attOff;
  float sum = 0.f;
  for (int c = 0; c < HH; ++c) sum += xp[c] * ap[c];
  s[idx] = sum;
}

__global__ void count_kernel(const int* __restrict__ src, const int* __restrict__ dst,
                             int* __restrict__ Dcnt, int* __restrict__ Bcnt, int nnz) {
  int i = blockIdx.x * 256 + threadIdx.x;
  if (i >= nnz) return;
  atomicAdd(&Dcnt[src[i]], 1);
  atomicAdd(&Bcnt[dst[i]], 1);
}

__global__ void alpha_max_kernel(const float* __restrict__ s1, const float* __restrict__ s2,
                                 const int* __restrict__ src, const int* __restrict__ dst,
                                 float* __restrict__ alpha, unsigned* __restrict__ segmax,
                                 int nnz, int heads) {
  int idx = blockIdx.x * 256 + threadIdx.x;
  if (idx >= nnz * heads) return;
  int i = idx / heads, h = idx % heads;
  float a = s1[src[i] * heads + h] + s2[dst[i] * heads + h];
  a = leaky(a, 0.2f);
  alpha[idx] = a;
  atomicMax(&segmax[src[i] * heads + h], fenc(a));
}

__global__ void alpha_exp_kernel(float* __restrict__ alpha, const unsigned* __restrict__ segmax,
                                 float* __restrict__ segsum, const int* __restrict__ src,
                                 int nnz, int heads) {
  int idx = blockIdx.x * 256 + threadIdx.x;
  if (idx >= nnz * heads) return;
  int i = idx / heads, h = idx % heads;
  float ex = __expf(alpha[idx] - fdec(segmax[src[i] * heads + h]));
  alpha[idx] = ex;
  atomicAdd(&segsum[src[i] * heads + h], ex);
}

__global__ void alpha_norm_kernel(float* __restrict__ alpha, const float* __restrict__ segsum,
                                  const int* __restrict__ src, int nnz, int heads) {
  int idx = blockIdx.x * 256 + threadIdx.x;
  if (idx >= nnz * heads) return;
  int i = idx / heads, h = idx % heads;
  alpha[idx] /= (segsum[src[i] * heads + h] + 1e-16f);
}

__global__ void eout_kernel(const float* __restrict__ xw, const float* __restrict__ alpha,
                            const int* __restrict__ src, const int* __restrict__ dst,
                            const int* __restrict__ Bcnt, float* __restrict__ eout,
                            int nnz, int heads) {
  int idx = blockIdx.x * 256 + threadIdx.x;
  int per = heads * HH;
  if (idx >= nnz * per) return;
  int i = idx / per, rem = idx % per;
  int h = rem >> 7, c = rem & 127;
  int d = dst[i];
  float binv = Bcnt[d] > 0 ? 1.f / (float)Bcnt[d] : 0.f;
  atomicAdd(&eout[((size_t)d * heads + h) * HH + c],
            binv * alpha[(size_t)i * heads + h] * xw[((size_t)src[i] * heads + h) * HH + c]);
}

__global__ void nodeagg_kernel(const float* __restrict__ eout, const float* __restrict__ alpha,
                               const int* __restrict__ src, const int* __restrict__ dst,
                               const int* __restrict__ Dcnt, float* __restrict__ outacc,
                               int nnz, int heads) {
  int idx = blockIdx.x * 256 + threadIdx.x;
  int per = heads * HH;
  if (idx >= nnz * per) return;
  int i = idx / per, rem = idx % per;
  int h = rem >> 7, c = rem & 127;
  int s = src[i];
  float dinv = Dcnt[s] > 0 ? 1.f / (float)Dcnt[s] : 0.f;
  atomicAdd(&outacc[((size_t)s * heads + h) * HH + c],
            dinv * alpha[(size_t)i * heads + h] * eout[((size_t)dst[i] * heads + h) * HH + c]);
}

__global__ void head_mean_kernel(const float* __restrict__ outacc, const float* __restrict__ bias,
                                 float* __restrict__ y, int n, int heads) {
  int idx = blockIdx.x * 256 + threadIdx.x;
  if (idx >= n * HH) return;
  int r = idx >> 7, c = idx & 127;
  float s = 0.f;
  for (int h = 0; h < heads; ++h) s += outacc[((size_t)r * heads + h) * HH + c];
  y[idx] = leaky(s / heads + bias[c], 0.2f);
}

__global__ void final_kernel(const float* __restrict__ x, const float* __restrict__ Wo,
                             const float* __restrict__ bo, float* __restrict__ out, int n) {
  int i = blockIdx.x * 256 + threadIdx.x;
  if (i >= n) return;
  float s = 0.f;
  for (int c = 0; c < HH; ++c) s += x[(size_t)i * HH + c] * Wo[c];
  out[i] = leaky(s + bo[0], 0.01f);
}

// ---------------- host side ----------------
extern "C" void kernel_launch(void* const* d_in, const int* in_sizes, int n_in,
                              void* d_out, int out_size, void* d_ws, size_t ws_size,
                              hipStream_t stream) {
  const float* price = (const float*)d_in[0];
  const int*   e     = (const int*)d_in[1];
  const float* bn1_g = (const float*)d_in[2];
  const float* bn1_b = (const float*)d_in[3];
  const float* W_ih  = (const float*)d_in[4];
  const float* W_hh  = (const float*)d_in[5];
  const float* b_ih  = (const float*)d_in[6];
  const float* b_hh  = (const float*)d_in[7];
  const float* Wq    = (const float*)d_in[8];
  const float* Wout  = (const float*)d_in[9];
  const float* ae    = (const float*)d_in[10];
  const float* ab    = (const float*)d_in[11];
  const float* bn2_g = (const float*)d_in[12];
  const float* bn2_b = (const float*)d_in[13];
  const float* W1    = (const float*)d_in[14];
  const float* att1  = (const float*)d_in[15];
  const float* bias1 = (const float*)d_in[16];
  const float* W2    = (const float*)d_in[17];
  const float* att2  = (const float*)d_in[18];
  const float* bias2 = (const float*)d_in[19];
  const float* Wo    = (const float*)d_in[20];
  const float* bo    = (const float*)d_in[21];

  const int* src = e;          // e[0]: node ids
  const int* dst = e + NNZV;   // e[1]: hyperedge ids

  float* out_nf = (float*)d_out;
  float* out_x  = out_nf + (size_t)NN * HH;
  float* out_o  = out_x  + (size_t)NN * HH;

  float* ws = (float*)d_ws;
  float* mean1 = ws;                 ws += 512;
  float* rstd1 = ws;                 ws += 512;
  float* mean2 = ws;                 ws += 128;
  float* rstd2 = ws;                 ws += 128;
  float* hT    = ws;                 ws += (size_t)NN * HH;
  float* ctx   = ws;                 // N*T*H floats — recycled after attention
  float* xn2    = ctx;
  float* ef     = xn2 + (size_t)NN * HH;
  float* xw     = ef + (size_t)EE * HH;
  float* ew     = xw + (size_t)NN * NHEADS * HH;
  float* s1     = ew + (size_t)EE * NHEADS * HH;
  float* s2     = s1 + (size_t)NN * NHEADS;
  float* alpha  = s2 + (size_t)EE * NHEADS;
  unsigned* segmax = (unsigned*)(alpha + (size_t)NNZV * NHEADS);
  float* segsum = (float*)(segmax + (size_t)NN * NHEADS);
  int*   Dcnt   = (int*)(segsum + (size_t)NN * NHEADS);
  int*   Bcnt   = Dcnt + NN;
  float* eout   = (float*)(Bcnt + EE);
  float* outacc = eout + (size_t)EE * NHEADS * HH;
  float* x2     = outacc + (size_t)NN * NHEADS * HH;
  float* ef2    = x2 + (size_t)NN * HH;
  float* xw2    = ef2 + (size_t)EE * HH;
  float* ew2    = xw2 + (size_t)NN * HH;
  float* alpha2 = ew2 + (size_t)EE * HH;
  unsigned* segmax2 = (unsigned*)(alpha2 + NNZV);
  float* segsum2 = (float*)(segmax2 + NN);

  // Stage 1: BN1 stats
  bn_stats_kernel<<<TT * FF, 256, 0, stream>>>(price, NN, TT * FF, mean1, rstd1);

  // Stage 2: fused GRU scan
  size_t gruLds = 185344;
  gru_kernel<<<NN / 16, 32, gruLds, stream>>>(price, mean1, rstd1, bn1_g, bn1_b,
                                              W_ih, W_hh, b_ih, b_hh, ctx, hT);

  // Stage 3: fused attention -> nf
  attn_kernel<<<NN / 16, 32, 0, stream>>>(hT, ctx, Wq, Wout, ae, ab, out_nf);

  // Stage 4: BN2
  bn_stats_kernel<<<HH, 256, 0, stream>>>(out_nf, NN, HH, mean2, rstd2);
  bn_apply_kernel<<<(NN * HH + 255) / 256, 256, 0, stream>>>(out_nf, mean2, rstd2,
                                                             bn2_g, bn2_b, xn2, NN * HH, HH);

  // Stage 5: hconv layer 1 (heads=4)
  hipMemsetAsync(ef,     0, (size_t)EE * HH * 4, stream);
  hipMemsetAsync(segmax, 0, (size_t)NN * NHEADS * 4, stream);
  hipMemsetAsync(segsum, 0, (size_t)NN * NHEADS * 4, stream);
  hipMemsetAsync(Dcnt,   0, (size_t)NN * 4, stream);
  hipMemsetAsync(Bcnt,   0, (size_t)EE * 4, stream);
  hipMemsetAsync(eout,   0, (size_t)EE * NHEADS * HH * 4, stream);
  hipMemsetAsync(outacc, 0, (size_t)NN * NHEADS * HH * 4, stream);

  edge_feat_kernel<<<(NNZV * HH) / 256, 256, 0, stream>>>(xn2, src, dst, ef, NNZV);
  gemm_wmma_kernel<<<dim3(NN / 16, (NHEADS * HH) / 64), 128, 0, stream>>>(xn2, W1, xw, NHEADS * HH);
  gemm_wmma_kernel<<<dim3(EE / 16, (NHEADS * HH) / 64), 128, 0, stream>>>(ef,  W1, ew, NHEADS * HH);
  att_score_kernel<<<(NN * NHEADS + 255) / 256, 256, 0, stream>>>(xw, att1, s1, NN, NHEADS, 0);
  att_score_kernel<<<(EE * NHEADS + 255) / 256, 256, 0, stream>>>(ew, att1, s2, EE, NHEADS, HH);
  count_kernel<<<NNZV / 256, 256, 0, stream>>>(src, dst, Dcnt, Bcnt, NNZV);
  alpha_max_kernel<<<(NNZV * NHEADS) / 256, 256, 0, stream>>>(s1, s2, src, dst, alpha, segmax, NNZV, NHEADS);
  alpha_exp_kernel<<<(NNZV * NHEADS) / 256, 256, 0, stream>>>(alpha, segmax, segsum, src, NNZV, NHEADS);
  alpha_norm_kernel<<<(NNZV * NHEADS) / 256, 256, 0, stream>>>(alpha, segsum, src, NNZV, NHEADS);
  eout_kernel<<<(NNZV * NHEADS * HH) / 256, 256, 0, stream>>>(xw, alpha, src, dst, Bcnt, eout, NNZV, NHEADS);
  nodeagg_kernel<<<(NNZV * NHEADS * HH) / 256, 256, 0, stream>>>(eout, alpha, src, dst, Dcnt, outacc, NNZV, NHEADS);
  head_mean_kernel<<<(NN * HH) / 256, 256, 0, stream>>>(outacc, bias1, x2, NN, NHEADS);

  // Stage 6: hconv layer 2 (heads=1)
  hipMemsetAsync(ef2,     0, (size_t)EE * HH * 4, stream);
  hipMemsetAsync(segmax2, 0, (size_t)NN * 4, stream);
  hipMemsetAsync(segsum2, 0, (size_t)NN * 4, stream);
  hipMemsetAsync(eout,    0, (size_t)EE * HH * 4, stream);
  hipMemsetAsync(outacc,  0, (size_t)NN * HH * 4, stream);

  edge_feat_kernel<<<(NNZV * HH) / 256, 256, 0, stream>>>(x2, src, dst, ef2, NNZV);
  gemm_wmma_kernel<<<dim3(NN / 16, HH / 64), 128, 0, stream>>>(x2,  W2, xw2, HH);
  gemm_wmma_kernel<<<dim3(EE / 16, HH / 64), 128, 0, stream>>>(ef2, W2, ew2, HH);
  att_score_kernel<<<(NN + 255) / 256, 256, 0, stream>>>(xw2, att2, s1, NN, 1, 0);
  att_score_kernel<<<(EE + 255) / 256, 256, 0, stream>>>(ew2, att2, s2, EE, 1, HH);
  alpha_max_kernel<<<NNZV / 256, 256, 0, stream>>>(s1, s2, src, dst, alpha2, segmax2, NNZV, 1);
  alpha_exp_kernel<<<NNZV / 256, 256, 0, stream>>>(alpha2, segmax2, segsum2, src, NNZV, 1);
  alpha_norm_kernel<<<NNZV / 256, 256, 0, stream>>>(alpha2, segsum2, src, NNZV, 1);
  eout_kernel<<<(NNZV * HH) / 256, 256, 0, stream>>>(xw2, alpha2, src, dst, Bcnt, eout, NNZV, 1);
  nodeagg_kernel<<<(NNZV * HH) / 256, 256, 0, stream>>>(eout, alpha2, src, dst, Dcnt, outacc, NNZV, 1);
  head_mean_kernel<<<(NN * HH) / 256, 256, 0, stream>>>(outacc, bias2, out_x, NN, 1);

  // Stage 7: final projection
  final_kernel<<<NN / 256, 256, 0, stream>>>(out_x, Wo, bo, out_o, NN);
}
